// SpKBGATConvOnly_59098749993533
// MI455X (gfx1250) — matrix operand order repaired and checked by reading, not compiled, not run on hardware
//
#include <hip/hip_runtime.h>
#include <hip/hip_bf16.h>
#include <math.h>

typedef __attribute__((ext_vector_type(16))) _Float16 v16h;
typedef __attribute__((ext_vector_type(8)))  _Float16 v8h;
typedef __attribute__((ext_vector_type(8)))  float    v8f;

#define NUM_REL   237
#define DDIM      200
#define CDIM      50
#define BATCH     8192
#define MAX_TILES 1280          // true bound: 237 + 8192/8 = 1261
#define WTK       224           // padded K extent of Wt
#define WTN       224           // padded N extent of Wt (only 0..207 read)

// ---- workspace layout (bytes) ----
#define WS_COUNTS   0
#define WS_OFFSETS  1024
#define WS_CURSORS  2048
#define WS_TILECNT  3072
#define WS_TILES    4096                        // 4 ints * 1280 tiles
#define WS_ROWLIST  (4096 + MAX_TILES*16)       // 24576
#define WS_SRC      (WS_ROWLIST + BATCH*4)      // 57344
#define WS_TAIL     (WS_SRC + BATCH*DDIM*4)     // 6610944
#define WS_WT       (WS_TAIL + BATCH*DDIM*4)    // 13164544 (1KB-aligned)
#define WS_DUMP     (WS_WT + NUM_REL*WTN*WTK*2) // 36947968 ; 1KB dump sink
// total ~36.95 MB

#define AS 232   // A-panel LDS row stride in halves (16B-aligned rows)

__global__ void zero_meta_k(int* counts, int* cursors, int* tilecnt) {
    int t = threadIdx.x;
    counts[t] = 0;
    cursors[t] = 0;
    if (t == 0) tilecnt[0] = 0;
}

__global__ void count_k(const long long* __restrict__ binp, int* counts) {
    int b = blockIdx.x * blockDim.x + threadIdx.x;
    if (b < BATCH) {
        int r = (int)binp[(size_t)b * 3 + 1];
        atomicAdd(&counts[r], 1);
    }
}

__global__ void scan_k(const int* __restrict__ counts, int* offsets) {
    if (threadIdx.x == 0) {
        int run = 0;
        for (int i = 0; i < NUM_REL; ++i) { offsets[i] = run; run += counts[i]; }
        offsets[NUM_REL] = run;
    }
}

__global__ void scatter_k(const long long* __restrict__ binp,
                          const int* __restrict__ offsets,
                          int* cursors, int* rowlist) {
    int b = blockIdx.x * blockDim.x + threadIdx.x;
    if (b < BATCH) {
        int r = (int)binp[(size_t)b * 3 + 1];
        int pos = offsets[r] + atomicAdd(&cursors[r], 1);
        rowlist[pos] = b;
    }
}

__global__ void tiles_k(const int* __restrict__ counts,
                        const int* __restrict__ offsets,
                        int* tilecnt, int* tiles) {
    int r = threadIdx.x;
    if (r < NUM_REL) {
        int n = counts[r];
        int off = offsets[r];
        int nt = (n + 7) >> 3;               // 8 triples -> 16 WMMA rows per tile
        for (int i = 0; i < nt; ++i) {
            int idx = atomicAdd(tilecnt, 1);
            tiles[idx * 4 + 0] = r;
            tiles[idx * 4 + 1] = off + i * 8;
            tiles[idx * 4 + 2] = min(8, n - i * 8);
        }
    }
}

// Pre-pass: Wt[r][n][k] = (f16) W[r][k][n], zero-padded to 224x224.
// LDS-tiled 32x32 transpose; coalesced reads and writes.
__global__ void __launch_bounds__(256)
wtrans_k(const float* __restrict__ W, _Float16* __restrict__ Wt) {
    const int r  = blockIdx.x / 7;
    const int nc = blockIdx.x % 7;           // 32-wide N chunk
    const float* Wr = W + (size_t)r * DDIM * DDIM;
    _Float16* WtR = Wt + (size_t)r * WTN * WTK;

    __shared__ float tile[32][33];
    const int tn = threadIdx.x & 31;
    const int tk = threadIdx.x >> 5;         // 0..7

    for (int kc = 0; kc < 7; ++kc) {
        const int n = nc * 32 + tn;
#pragma unroll
        for (int i = 0; i < 4; ++i) {
            int k = kc * 32 + tk + 8 * i;
            float v = (k < DDIM && n < DDIM) ? Wr[(size_t)k * DDIM + n] : 0.0f;
            tile[tk + 8 * i][tn] = v;        // tile[k_local][n_local]
        }
        __syncthreads();
#pragma unroll
        for (int i = 0; i < 4; ++i) {
            int nn = tk + 8 * i;             // n_local
            WtR[(size_t)(nc * 32 + nn) * WTK + kc * 32 + tn] =
                (_Float16)tile[tn][nn];      // k_local = tn (conflict-free, stride 33)
        }
        __syncthreads();
    }
}

__device__ __forceinline__ float fast_tanh(float x) {
    // tanh(x) = (e - 1) * rcp(e + 1), e = 2^(2*log2(e)*x); clamp handles saturation
    x = fminf(fmaxf(x, -15.0f), 15.0f);
    float e = __builtin_amdgcn_exp2f(x * 2.8853900817779268f);
    return (e - 1.0f) * __builtin_amdgcn_rcpf(e + 1.0f);
}

// One block = one tile: 16 rows (8 triples x {head,tail}) x D outputs, GEMM vs Wt_r.
// 4 waves; wave w covers N-tiles {w, w+4, w+8, w+12} of the 13 N-tiles.
__global__ void __launch_bounds__(128)
gemm_tanh_k(const long long* __restrict__ binp,
            const float* __restrict__ entEmb,
            const _Float16* __restrict__ Wt,
            const int* __restrict__ tilecnt,
            const int* __restrict__ tiles,
            const int* __restrict__ rowlist,
            float* __restrict__ SRC, float* __restrict__ TAIL,
            float* __restrict__ dump) {
    int tc = tilecnt[0];
    if ((int)blockIdx.x >= tc) return;

    __shared__ _Float16 Alds[16][AS];
    __shared__ int eidx[16];
    __shared__ int bArr[8];

    const int r     = tiles[blockIdx.x * 4 + 0];
    const int start = tiles[blockIdx.x * 4 + 1];
    const int cnt   = tiles[blockIdx.x * 4 + 2];

    const int t = threadIdx.x;
    // row m = 2i -> head of triple i ; row m = 2i+1 -> tail of triple i
    if (t < 16) {
        int i = t >> 1;
        if (i < cnt) {
            int b = rowlist[start + i];
            if ((t & 1) == 0) bArr[i] = b;
            long long e = binp[(size_t)b * 3 + ((t & 1) ? 2 : 0)];
            eidx[t] = (int)e;
        } else {
            eidx[t] = -1;
            if ((t & 1) == 0) bArr[i] = 0;
        }
    }
    __syncthreads();

    const _Float16* WtR = Wt + (size_t)r * WTN * WTK;

    // warm L2/L1 with this relation's Wt block (global_prefetch_b8)
    {
        const char* wb = (const char*)WtR;
        for (size_t off = (size_t)t * 64; off < (size_t)WTN * WTK * 2; off += 128 * 64)
            __builtin_prefetch(wb + off, 0, 1);
    }

    // stage gathered A panel (fp32 -> f16), zero-padded rows/cols
    for (int f = t; f < 16 * AS; f += 128) {
        int m = f / AS, k = f % AS;
        int e = eidx[m];
        float v = (e >= 0 && k < DDIM) ? entEmb[(size_t)e * DDIM + k] : 0.0f;
        Alds[m][k] = (_Float16)v;
    }
    __syncthreads();

    const int wave = t >> 5;
    const int l    = t & 31;
    const int mrow = l & 15;
    const int koff = (l < 16) ? 0 : 16;  // A layout: lanes16-31 hold K{16-31}
    const int ncol = l & 15;
    const int boff = (l & 16) ? 16 : 0;  // B layout: lanes16-31 hold K+16

    // preload all 7 A fragments once (shared across every N-tile)
    v16h afr[7];
#pragma unroll
    for (int kt = 0; kt < 7; ++kt) {
        const _Float16* ap = &Alds[mrow][kt * 32 + koff];
        v8h a0 = *(const v8h*)ap;
        v8h a1 = *(const v8h*)(ap + 8);
        v16h af;
#pragma unroll
        for (int i = 0; i < 8; ++i) { af[i] = a0[i]; af[i + 8] = a1[i]; }
        afr[kt] = af;
    }

    // hoisted branch-free epilogue descriptors (loop-invariant per lane)
    float* baseJ[8];
    bool   okJ[8];
#pragma unroll
    for (int j = 0; j < 8; ++j) {
        int m = j + ((l & 16) ? 8 : 0);      // C/D layout: M = j + 8*(lane>=16)
        int i = m >> 1;
        okJ[j]   = (i < cnt);
        baseJ[j] = ((m & 1) ? TAIL : SRC) + (size_t)bArr[i] * DDIM;
    }

    for (int nt = wave; nt < 13; nt += 4) {
        const int n = nt * 16 + ncol;               // <= 207, always within Wt pad
        const _Float16* wrow = WtR + (size_t)n * WTK;
        const bool nin = (n < DDIM);

        // software pipeline: issue all 14 b128 loads up front (max MLP),
        // then run the WMMA chain with staggered waits.
        v16h bfr[7];
#pragma unroll
        for (int kt = 0; kt < 7; ++kt)
            bfr[kt] = *(const v16h*)(wrow + kt * 32 + boff);

        v8f acc = {0.f, 0.f, 0.f, 0.f, 0.f, 0.f, 0.f, 0.f};
#pragma unroll
        for (int kt = 0; kt < 7; ++kt)
            acc = __builtin_amdgcn_wmma_f32_16x16x32_f16(
                false, afr[kt], false, bfr[kt], (short)0, acc, false, false);

        // branch-free epilogue: fast tanh + pointer-select scatter
#pragma unroll
        for (int j = 0; j < 8; ++j) {
            float y = fast_tanh(acc[j]);
            float* p = (nin && okJ[j]) ? (baseJ[j] + n) : dump;
            *p = y;
        }
    }
}

// conv (3-tap over {src',rel,tail'}) + relu + fc dot. One wave per triple.
__global__ void __launch_bounds__(256)
convfc_k(const long long* __restrict__ binp,
         const float* __restrict__ relEmb,
         const float* __restrict__ convw, const float* __restrict__ convb,
         const float* __restrict__ fcw, const float* __restrict__ fcb,
         const float* __restrict__ SRC, const float* __restrict__ TAIL,
         float* __restrict__ out) {
    __shared__ float s_fc[CDIM * DDIM];
    __shared__ float s_cw[CDIM * 3];
    __shared__ float s_cb[CDIM];

    for (int f = threadIdx.x; f < CDIM * DDIM; f += 256) s_fc[f] = fcw[f];
    if (threadIdx.x < CDIM * 3) s_cw[threadIdx.x] = convw[threadIdx.x];
    if (threadIdx.x < CDIM)     s_cb[threadIdx.x] = convb[threadIdx.x];
    __syncthreads();

    const int wave = threadIdx.x >> 5;
    const int l    = threadIdx.x & 31;
    const int b    = blockIdx.x * 8 + wave;
    if (b >= BATCH) return;

    const int r = (int)binp[(size_t)b * 3 + 1];
    float s[7], rr[7], tt[7];
#pragma unroll
    for (int j = 0; j < 7; ++j) {
        int d = l + 32 * j;
        if (d < DDIM) {
            s[j]  = SRC[(size_t)b * DDIM + d];
            rr[j] = relEmb[(size_t)r * DDIM + d];
            tt[j] = TAIL[(size_t)b * DDIM + d];
        } else { s[j] = 0.f; rr[j] = 0.f; tt[j] = 0.f; }
    }

    float acc = 0.f;
    for (int c = 0; c < CDIM; ++c) {
        float w0 = s_cw[c * 3 + 0], w1 = s_cw[c * 3 + 1], w2 = s_cw[c * 3 + 2];
        float cb = s_cb[c];
#pragma unroll
        for (int j = 0; j < 7; ++j) {
            int d = l + 32 * j;
            if (d < DDIM) {
                float v = fmaf(w0, s[j], fmaf(w1, rr[j], fmaf(w2, tt[j], cb)));
                v = fmaxf(v, 0.0f);
                acc = fmaf(v, s_fc[c * DDIM + d], acc);
            }
        }
    }
    for (int o = 16; o > 0; o >>= 1) acc += __shfl_down(acc, o, 32);
    if (l == 0) out[b] = acc + fcb[0];
}

extern "C" void kernel_launch(void* const* d_in, const int* in_sizes, int n_in,
                              void* d_out, int out_size, void* d_ws, size_t ws_size,
                              hipStream_t stream) {
    const long long* binp   = (const long long*)d_in[0];  // (B,3) int64
    const float*     entEmb = (const float*)d_in[1];      // (100000,200)
    const float*     relEmb = (const float*)d_in[2];      // (237,200)
    const float*     W      = (const float*)d_in[3];      // (237,200,200)
    const float*     convw  = (const float*)d_in[4];      // (50,3)
    const float*     convb  = (const float*)d_in[5];      // (50,)
    const float*     fcw    = (const float*)d_in[6];      // (1,10000)
    const float*     fcb    = (const float*)d_in[7];      // (1,)
    float* out = (float*)d_out;

    char* ws = (char*)d_ws;
    int* counts  = (int*)(ws + WS_COUNTS);
    int* offsets = (int*)(ws + WS_OFFSETS);
    int* cursors = (int*)(ws + WS_CURSORS);
    int* tilecnt = (int*)(ws + WS_TILECNT);
    int* tiles   = (int*)(ws + WS_TILES);
    int* rowlist = (int*)(ws + WS_ROWLIST);
    float* SRC   = (float*)(ws + WS_SRC);
    float* TAIL  = (float*)(ws + WS_TAIL);
    _Float16* Wt = (_Float16*)(ws + WS_WT);
    float* dump  = (float*)(ws + WS_DUMP);

    // independent pre-pass + binning pipeline
    wtrans_k<<<NUM_REL * 7, 256, 0, stream>>>(W, Wt);
    zero_meta_k<<<1, 256, 0, stream>>>(counts, cursors, tilecnt);
    count_k<<<(BATCH + 255) / 256, 256, 0, stream>>>(binp, counts);
    scan_k<<<1, 64, 0, stream>>>(counts, offsets);
    scatter_k<<<(BATCH + 255) / 256, 256, 0, stream>>>(binp, offsets, cursors, rowlist);
    tiles_k<<<1, 256, 0, stream>>>(counts, offsets, tilecnt, tiles);
    gemm_tanh_k<<<MAX_TILES, 128, 0, stream>>>(binp, entEmb, Wt, tilecnt, tiles,
                                               rowlist, SRC, TAIL, dump);
    convfc_k<<<BATCH / 8, 256, 0, stream>>>(binp, relEmb, convw, convb, fcw, fcb,
                                            SRC, TAIL, out);
}
